// RegularHMM_39032662786340
// MI455X (gfx1250) — matrix operand order repaired and checked by reading, not compiled, not run on hardware
//
#include <hip/hip_runtime.h>

typedef _Float16 v16h __attribute__((ext_vector_type(16)));
typedef _Float16 v8h  __attribute__((ext_vector_type(8)));
typedef float    v8f  __attribute__((ext_vector_type(8)));

static constexpr int KS = 512;        // states
static constexpr int TS = 256;        // time steps
static constexpr int ASTRIDE = 520;   // halfs per LDS A row (1040B row stride -> conflict-free b128)

// B fragment storage layout (both global ws and LDS):
//   plane p = kb*4 + khalf*2 + j   (j = 0/1 selects K sub-chunk of 8)
//   halfs  trans[(p*512 + n)*8 + pos],  K = kb*32 + khalf*16 + j*8 + pos
// A lane's b128 read is at byte (p*512+n)*16: lane stride 16B = 4 dwords -> perfect bank coverage.
static constexpr size_t B_LDS_BYTES = (size_t)8 * 4 * 512 * 16;  // kb 0..7 cached: 262144
static constexpr size_t APOST_OFF   = B_LDS_BYTES;               // 16*520*2 = 16640
static constexpr size_t XT_OFF      = APOST_OFF + 16 * ASTRIDE * 2;   // 278784
static constexpr size_t PART_OFF    = XT_OFF + 64;                    // 16 waves * 16 floats
static constexpr size_t RECIP_OFF   = PART_OFF + 16 * 16 * 4;
static constexpr size_t SMEM_BYTES  = RECIP_OFF + 64;                 // 279936 < 320KB

// ---------------- Prep 1: transition row softmax -> f16, planed B-fragment layout ----------------
__global__ __launch_bounds__(256) void hmm_prep_trans(const float* __restrict__ traw,
                                                      _Float16* __restrict__ trans) {
    __shared__ float red[256];
    const int k   = blockIdx.x;
    const int tid = threadIdx.x;
    float v0 = traw[k * KS + tid];
    float v1 = traw[k * KS + tid + 256];
    float m  = fmaxf(v0, v1);
    red[tid] = m; __syncthreads();
    for (int s = 128; s > 0; s >>= 1) { if (tid < s) red[tid] = fmaxf(red[tid], red[tid + s]); __syncthreads(); }
    m = red[0]; __syncthreads();
    float e0 = __expf(v0 - m), e1 = __expf(v1 - m);
    red[tid] = e0 + e1; __syncthreads();
    for (int s = 128; s > 0; s >>= 1) { if (tid < s) red[tid] += red[tid + s]; __syncthreads(); }
    float rs = 1.0f / red[0];

    const int kb = k >> 5, kr = k & 31;
    const int kh = kr >> 4, j = (kr >> 3) & 1, pos = kr & 7;
    const size_t plane = (size_t)(kb * 4 + kh * 2 + j);
    trans[(plane * 512 + tid      ) * 8 + pos] = (_Float16)(e0 * rs);
    trans[(plane * 512 + tid + 256) * 8 + pos] = (_Float16)(e1 * rs);
}

// ---------------- Prep 2: emission softmax + exp(prior) ----------------
__global__ __launch_bounds__(256) void hmm_prep_small(const float* __restrict__ eraw,
                                                      const float* __restrict__ praw,
                                                      float* __restrict__ em0,
                                                      float* __restrict__ em1,
                                                      float* __restrict__ pe) {
    int k = blockIdx.x * blockDim.x + threadIdx.x;
    if (k < KS) {
        float a = eraw[2 * k], b = eraw[2 * k + 1];
        float m = fmaxf(a, b);
        float x0 = __expf(a - m), x1 = __expf(b - m);
        float r = 1.0f / (x0 + x1);
        em0[k] = x0 * r;
        em1[k] = x1 * r;
        pe[k]  = __expf(praw[k]);   // prior's global softmax denom cancels in row-normalize
    }
}

__device__ __forceinline__ float wave_xor_add16(float s) {
    s += __int_as_float(__builtin_amdgcn_ds_swizzle(__float_as_int(s), 0x041f));
    s += __int_as_float(__builtin_amdgcn_ds_swizzle(__float_as_int(s), 0x081f));
    s += __int_as_float(__builtin_amdgcn_ds_swizzle(__float_as_int(s), 0x101f));
    s += __int_as_float(__builtin_amdgcn_ds_swizzle(__float_as_int(s), 0x201f));
    return s;
}

__device__ __forceinline__ v16h mkfrag(v8h a, v8h b) {
    return __builtin_shufflevector(a, b, 0,1,2,3,4,5,6,7,8,9,10,11,12,13,14,15);
}

// ---------------- Main: 16-row HMM recursion per WG, 16 waves, LDS-cached + streamed B ----------------
__global__ __launch_bounds__(512) void hmm_main(const float* __restrict__ x,       // [256][256]
                                                const _Float16* __restrict__ trans,
                                                const float* __restrict__ em0g,
                                                const float* __restrict__ em1g,
                                                const float* __restrict__ peg,
                                                float* __restrict__ out) {         // [256][512]
    extern __shared__ char smem[];
    _Float16* Btile   = (_Float16*)smem;                 // planes 0..31 (kb 0..7), 256KB
    _Float16* Apost   = (_Float16*)(smem + APOST_OFF);
    float*    xt      = (float*)(smem + XT_OFF);
    float*    partial = (float*)(smem + PART_OFF);       // [16][16]
    float*    recip   = (float*)(smem + RECIP_OFF);

    const int tid   = threadIdx.x;
    const int lane  = tid & 31;
    const int wv    = tid >> 5;          // 0..15
    const int l16   = lane & 15;
    const int khalf = lane >> 4;
    const int r0    = blockIdx.x * 16;   // batch rows of this WG
    const int nbase = wv * 32;           // this wave's 32-column slab (2 tiles)

    // ---- one-time: async-copy first half of transition (kb 0..7) into LDS ----
    {
#pragma unroll 4
        for (int c = 0; c < 32; ++c) {
            const int idx = (c * 512 + tid) * 16;                       // byte offset, 16B chunks
            uint32_t ldsa = (uint32_t)(uintptr_t)(smem + idx);
            uint64_t ga   = (uint64_t)(uintptr_t)((const char*)trans + idx);
            asm volatile("global_load_async_to_lds_b128 %0, %1, off"
                         :: "v"(ldsa), "v"(ga) : "memory");
        }
        asm volatile("s_wait_asynccnt 0x0" ::: "memory");
    }

    // per-lane per-tile constants (column n fixed for the whole recursion)
    int   ncol[2];
    float e0c[2], dec[2], pec[2];
#pragma unroll
    for (int nt = 0; nt < 2; ++nt) {
        int n = nbase + nt * 16 + l16;
        ncol[nt] = n;
        float a = em0g[n], b = em1g[n];
        e0c[nt] = a; dec[nt] = b - a;
        pec[nt] = peg[n];
    }

    if (wv == 0 && lane < 16) xt[lane] = x[(r0 + lane) * TS + 0];
    __syncthreads();    // publishes LDS B-tile copy + xt

    v8f acc[2];

    for (int t = 0; t < TS; ++t) {
        if (t == 0) {
#pragma unroll
            for (int nt = 0; nt < 2; ++nt)
#pragma unroll
                for (int i = 0; i < 8; ++i) acc[nt][i] = pec[nt];
        } else {
#pragma unroll
            for (int nt = 0; nt < 2; ++nt) {
                v8f z = {0.f, 0.f, 0.f, 0.f, 0.f, 0.f, 0.f, 0.f};
                acc[nt] = z;
            }
            // prefetch first global k-block (kb=8) while LDS half computes
            v16h bb[2][2];
#pragma unroll
            for (int nt = 0; nt < 2; ++nt) {
                const _Float16* p = &trans[(size_t)((8 * 4 + khalf * 2) * 512 + ncol[nt]) * 8];
                bb[0][nt] = mkfrag(*(const v8h*)p, *(const v8h*)(p + 512 * 8));
            }
            // ---- LDS half: kb 0..7, B from LDS (conflict-free ds_load_b128) ----
            for (int kb = 0; kb < 8; ++kb) {
                const _Float16* ap = &Apost[l16 * ASTRIDE + kb * 32 + khalf * 16];
                v16h a = mkfrag(*(const v8h*)ap, *(const v8h*)(ap + 8));
#pragma unroll
                for (int nt = 0; nt < 2; ++nt) {
                    const _Float16* bp = &Btile[(size_t)((kb * 4 + khalf * 2) * 512 + ncol[nt]) * 8];
                    v16h b = mkfrag(*(const v8h*)bp, *(const v8h*)(bp + 512 * 8));
                    acc[nt] = __builtin_amdgcn_wmma_f32_16x16x32_f16(
                        false, a, false, b, (short)0, acc[nt], false, false);
                }
            }
            // ---- streamed half: kb 8..15, B from L2 with double-buffered prefetch ----
            for (int kb = 8; kb < 16; ++kb) {
                const int cb = kb & 1;
                if (kb < 15) {
#pragma unroll
                    for (int nt = 0; nt < 2; ++nt) {
                        const _Float16* p =
                            &trans[(size_t)(((kb + 1) * 4 + khalf * 2) * 512 + ncol[nt]) * 8];
                        bb[cb ^ 1][nt] = mkfrag(*(const v8h*)p, *(const v8h*)(p + 512 * 8));
                    }
                }
                const _Float16* ap = &Apost[l16 * ASTRIDE + kb * 32 + khalf * 16];
                v16h a = mkfrag(*(const v8h*)ap, *(const v8h*)(ap + 8));
                acc[0] = __builtin_amdgcn_wmma_f32_16x16x32_f16(
                    false, a, false, bb[cb][0], (short)0, acc[0], false, false);
                acc[1] = __builtin_amdgcn_wmma_f32_16x16x32_f16(
                    false, a, false, bb[cb][1], (short)0, acc[1], false, false);
            }
        }

        // evidence: acc[m,n] *= em0[n] + x[m,t]*(em1[n]-em0[n]);  m = i + 8*khalf
        float xr[8];
#pragma unroll
        for (int i = 0; i < 8; ++i) xr[i] = xt[i + 8 * khalf];
#pragma unroll
        for (int nt = 0; nt < 2; ++nt)
#pragma unroll
            for (int i = 0; i < 8; ++i)
                acc[nt][i] *= fmaf(xr[i], dec[nt], e0c[nt]);

        // row sums: cross-lane xor reduction, per-wave partials
#pragma unroll
        for (int i = 0; i < 8; ++i) {
            float s = acc[0][i] + acc[1][i];
            s = wave_xor_add16(s);
            if (l16 == 0) partial[wv * 16 + i + 8 * khalf] = s;
        }
        __syncthreads();

        if (wv == 0 && lane < 16) {
            float tot = 0.f;
#pragma unroll
            for (int w = 0; w < 16; ++w) tot += partial[w * 16 + lane];
            recip[lane] = 1.0f / tot;
        }
        if (wv == 1 && lane < 16 && t + 1 < TS)   // prefetch next step's evidence inputs
            xt[lane] = x[(r0 + lane) * TS + t + 1];
        __syncthreads();

        float rr[8];
#pragma unroll
        for (int i = 0; i < 8; ++i) rr[i] = recip[i + 8 * khalf];

        if (t + 1 < TS) {
#pragma unroll
            for (int nt = 0; nt < 2; ++nt)
#pragma unroll
                for (int i = 0; i < 8; ++i)
                    Apost[(i + 8 * khalf) * ASTRIDE + ncol[nt]] = (_Float16)(acc[nt][i] * rr[i]);
        } else {
#pragma unroll
            for (int nt = 0; nt < 2; ++nt)
#pragma unroll
                for (int i = 0; i < 8; ++i)
                    out[(size_t)(r0 + i + 8 * khalf) * KS + ncol[nt]] = acc[nt][i] * rr[i];
        }
        __syncthreads();
    }
}

extern "C" void kernel_launch(void* const* d_in, const int* in_sizes, int n_in,
                              void* d_out, int out_size, void* d_ws, size_t ws_size,
                              hipStream_t stream) {
    const float* x    = (const float*)d_in[0];   // [256][256]
    const float* praw = (const float*)d_in[1];   // [512]
    const float* traw = (const float*)d_in[2];   // [512][512]
    const float* eraw = (const float*)d_in[3];   // [512][2]

    char* ws = (char*)d_ws;
    _Float16* trans = (_Float16*)ws;                       // 512*512*2 = 524288 B (planed layout)
    float* em0 = (float*)(ws + 524288);
    float* em1 = (float*)(ws + 524288 + 2048);
    float* pe  = (float*)(ws + 524288 + 4096);

    hmm_prep_trans<<<512, 256, 0, stream>>>(traw, trans);
    hmm_prep_small<<<2, 256, 0, stream>>>(eraw, praw, em0, em1, pe);
    hmm_main<<<16, 512, SMEM_BYTES, stream>>>(x, trans, em0, em1, pe, (float*)d_out);
}
